// HPLFlowNetShallow_68418829025449
// MI455X (gfx1250) — compile-verified, matrix operand order checked by reading, not run on hardware
//
#include <hip/hip_runtime.h>

// ============================================================================
// HPLFlowNetShallow on MI455X (gfx1250).
// All channel-mixing math funnels through one branch-free bf16 WMMA GEMM:
//   - W pre-packed to zero-padded bf16 (A-frag = two b128 loads / K-step)
//   - X buffers K-padded in workspace (unguarded coalesced dword loads)
//   - 32x32 output tile per wave: 2 A-frags x 2 B-frags -> 4 WMMAs / K-step
// Splat/gather/slice remain scalar memory kernels (atomic f32 scatter-add).
// ============================================================================

typedef __attribute__((ext_vector_type(16))) __bf16 v16bf;
typedef __attribute__((ext_vector_type(8)))  __bf16 v8bf;
typedef __attribute__((ext_vector_type(8)))  float  v8f;

#define NBATCH 4

__device__ __forceinline__ float lrelu_f(float x) { return x >= 0.f ? x : 0.1f * x; }

// ---------------------------------------------------------------------------
// Pack W [O,C] f32 -> Wp [Opad, Cs] bf16, zero padded (Opad mult 64, Cs mult 32)
// ---------------------------------------------------------------------------
__global__ __launch_bounds__(256) void pack_w_kernel(
    const float* __restrict__ W, __bf16* __restrict__ Wp,
    int O, int C, int Cs, long total)
{
  long id = (long)blockIdx.x * 256 + threadIdx.x;
  if (id >= total) return;
  int  k = (int)(id % Cs);
  long m = id / Cs;
  float v = (m < O && k < C) ? W[(size_t)m * C + k] : 0.f;
  Wp[id] = (__bf16)v;
}

// zero rows [C, Cpad) of buf [B, Cpad, n]
__global__ void zero_rows_kernel(float* __restrict__ buf,
                                 int C, int Cpad, int n, long total)
{
  long id = (long)blockIdx.x * 256 + threadIdx.x;
  if (id >= total) return;
  const int R = Cpad - C;
  int x = (int)(id % n);
  long t = id / n;
  int r = (int)(t % R);
  int b = (int)(t / R);
  buf[((size_t)b * Cpad + C + r) * n + x] = 0.f;
}

// ---------------------------------------------------------------------------
// Y[b,o,n] = act( sum_c Wp[o,c] * X[b,c,n] + bias[o] )
// Wp: bf16 [Opad,Cs] (padded, Cs mult 32). X: f32 [B,Cs,Nn] (pad rows zero),
// Nn mult 128. 8 waves/block: 2 M-waves x 4 N-waves, each wave 32(M)x32(N):
// 2 A-frags (b128 pairs) x 2 B-frags (strided dword + cvt) -> 4 WMMA / K-step.
// ---------------------------------------------------------------------------
__global__ __launch_bounds__(256) void gemm_wmma_kernel(
    const __bf16* __restrict__ Wp, const float* __restrict__ Bias,
    const float* __restrict__ X, float* __restrict__ Y,
    int O, int Cs, int Nn, int act)
{
  const int b    = blockIdx.z;
  const int lane = threadIdx.x & 31;
  const int wid  = threadIdx.x >> 5;
  const int m0   = blockIdx.y * 64  + (wid >> 2) * 32;
  const int n0   = blockIdx.x * 128 + (wid & 3) * 32;
  const int half = lane >> 4;
  const int l16  = lane & 15;
  const int n1   = n0 + l16;
  const int n2   = n1 + 16;
  const float*  Xb  = X  + (size_t)b * Cs * Nn;
  const __bf16* Wr0 = Wp + (size_t)(m0 + l16) * Cs + 8 * half;
  const __bf16* Wr1 = Wr0 + (size_t)16 * Cs;

  v8f acc00 = {}, acc01 = {}, acc10 = {}, acc11 = {};
  for (int k0 = 0; k0 < Cs; k0 += 32) {
    const v8bf lo0 = *(const v8bf*)(Wr0 + k0);
    const v8bf hi0 = *(const v8bf*)(Wr0 + k0 + 16);
    const v8bf lo1 = *(const v8bf*)(Wr1 + k0);
    const v8bf hi1 = *(const v8bf*)(Wr1 + k0 + 16);
    v16bf af0, af1;
#pragma unroll
    for (int i = 0; i < 8; ++i) {
      af0[i] = lo0[i]; af0[8 + i] = hi0[i];
      af1[i] = lo1[i]; af1[8 + i] = hi1[i];
    }
    v16bf bf1, bf2;
#pragma unroll
    for (int j = 0; j < 8; ++j) {
      const int k = k0 + 2 * (j & 3) + 16 * (j >> 2) + 8 * half;
      const float* r0 = Xb + (size_t)k * Nn;
      const float* r1 = r0 + Nn;
      bf1[2 * j]     = (__bf16)r0[n1];
      bf1[2 * j + 1] = (__bf16)r1[n1];
      bf2[2 * j]     = (__bf16)r0[n2];
      bf2[2 * j + 1] = (__bf16)r1[n2];
    }
    acc00 = __builtin_amdgcn_wmma_f32_16x16x32_bf16(false, af0, false, bf1, (short)0, acc00, false, false);
    acc01 = __builtin_amdgcn_wmma_f32_16x16x32_bf16(false, af0, false, bf2, (short)0, acc01, false, false);
    acc10 = __builtin_amdgcn_wmma_f32_16x16x32_bf16(false, af1, false, bf1, (short)0, acc10, false, false);
    acc11 = __builtin_amdgcn_wmma_f32_16x16x32_bf16(false, af1, false, bf2, (short)0, acc11, false, false);
  }
  float* Yb = Y + (size_t)b * O * Nn;
#pragma unroll
  for (int v = 0; v < 8; ++v) {
    const int mr0 = m0 + half * 8 + v;   // C/D layout: lanes 16-31 hold M+8
    const int mr1 = mr0 + 16;
    if (mr0 < O) {
      const float bi = Bias[mr0];
      const float y0 = acc00[v] + bi;
      const float y1 = acc01[v] + bi;
      Yb[(size_t)mr0 * Nn + n1] = act ? lrelu_f(y0) : y0;
      Yb[(size_t)mr0 * Nn + n2] = act ? lrelu_f(y1) : y1;
    }
    if (mr1 < O) {
      const float bi = Bias[mr1];
      const float y0 = acc10[v] + bi;
      const float y1 = acc11[v] + bi;
      Yb[(size_t)mr1 * Nn + n1] = act ? lrelu_f(y0) : y0;
      Yb[(size_t)mr1 * Nn + n2] = act ? lrelu_f(y1) : y1;
    }
  }
}

// ---------------------------------------------------------------------------
// splat: lat[b, c, off[b,d,p]] += src[b,c,p] * bary[b,d,p], d=0..3.
// ---------------------------------------------------------------------------
__global__ void splat_kernel(const float* __restrict__ el,
                             const float* __restrict__ feat,
                             const float* __restrict__ bary,
                             const int*   __restrict__ off,
                             float* __restrict__ lat,
                             int Cfeat, int hasEl, int npt, int h, long total)
{
  long id = (long)blockIdx.x * blockDim.x + threadIdx.x;
  if (id >= total) return;
  const int Ctot = Cfeat + (hasEl ? 4 : 0);
  int p = (int)(id % npt);
  long t = id / npt;
  int c = (int)(t % Ctot);
  int b = (int)(t / Ctot);
  float v = (hasEl && c < 4)
              ? el[((size_t)b * 4 + c) * npt + p]
              : feat[((size_t)b * Cfeat + (c - (hasEl ? 4 : 0))) * npt + p];
#pragma unroll
  for (int d = 0; d < 4; ++d) {
    float w = bary[((size_t)b * 4 + d) * npt + p];
    int   o = off [((size_t)b * 4 + d) * npt + p];
    atomicAdd(&lat[((size_t)b * Ctot + c) * h + o], v * w);
  }
}

// out[b, c*K+k, x] = lat[b, c, nb[b,k,x]], out batch stride CKpad*h
__global__ void gather_k_kernel(const float* __restrict__ lat,
                                const int* __restrict__ nb,
                                float* __restrict__ out,
                                int C, int K, int h, int CKpad, long total)
{
  long id = (long)blockIdx.x * blockDim.x + threadIdx.x;
  if (id >= total) return;
  int x = (int)(id % h);
  long t = id / h;
  int k = (int)(t % K); t /= K;
  int c = (int)(t % C);
  int b = (int)(t / C);
  int j = nb[((size_t)b * K + k) * h + x];
  out[((size_t)b * CKpad + c * K + k) * h + x] = lat[((size_t)b * C + c) * h + j];
}

// out[b,c,p] = sum_d lat[b,c,off[b,d,p]] * bary[b,d,p]
__global__ void slice_kernel(const float* __restrict__ lat,
                             const float* __restrict__ bary,
                             const int* __restrict__ off,
                             float* __restrict__ out,
                             int C, int h, int npt, long total)
{
  long id = (long)blockIdx.x * blockDim.x + threadIdx.x;
  if (id >= total) return;
  int p = (int)(id % npt);
  long t = id / npt;
  int c = (int)(t % C);
  int b = (int)(t / C);
  float s = 0.f;
#pragma unroll
  for (int d = 0; d < 4; ++d) {
    int   o = off [((size_t)b * 4 + d) * npt + p];
    float w = bary[((size_t)b * 4 + d) * npt + p];
    s += lat[((size_t)b * C + c) * h + o] * w;
  }
  out[((size_t)b * C + c) * npt + p] = s;
}

// dst[b, c0+c, x] = src[b, c, x]  (dst batch stride Cdst*n)
__global__ void copy_ch_kernel(const float* __restrict__ src,
                               float* __restrict__ dst,
                               int Csrc, int n, int Cdst, int c0, long total)
{
  long id = (long)blockIdx.x * blockDim.x + threadIdx.x;
  if (id >= total) return;
  int x = (int)(id % n);
  long t = id / n;
  int c = (int)(t % Csrc);
  int b = (int)(t / Csrc);
  dst[((size_t)b * Cdst + c0 + c) * n + x] = src[((size_t)b * Csrc + c) * n + x];
}

// prod[b, c, kk*h+x]:  c<C1 -> f1[b,c,x]*f2[b,c,idx2[b,kk,x]],
//                      c>=C1 -> prevlat[b,c-C1,x] (broadcast over kk)
__global__ void corr_prod_kernel(const float* __restrict__ f1,
                                 const float* __restrict__ f2,
                                 const int* __restrict__ idx2,
                                 const float* __restrict__ prevlat,
                                 float* __restrict__ prod,
                                 int C1, int Cp, int K2, int h, long total)
{
  long id = (long)blockIdx.x * blockDim.x + threadIdx.x;
  if (id >= total) return;
  const int Ct = C1 + Cp;
  int x = (int)(id % h);
  long t = id / h;
  int kk = (int)(t % K2); t /= K2;
  int c  = (int)(t % Ct);
  int b  = (int)(t / Ct);
  float v;
  if (c < C1) {
    int j = idx2[((size_t)b * K2 + kk) * h + x];
    v = f1[((size_t)b * C1 + c) * h + x] * f2[((size_t)b * C1 + c) * h + j];
  } else {
    v = prevlat[((size_t)b * Cp + (c - C1)) * h + x];
  }
  prod[(((size_t)b * Ct + c) * K2 + kk) * h + x] = v;
}

// out[b,o,x] = mean_k mid[b, o, k*h+x]
__global__ void mean_k_kernel(const float* __restrict__ mid,
                              float* __restrict__ out,
                              int O, int K, int h, long total)
{
  long id = (long)blockIdx.x * blockDim.x + threadIdx.x;
  if (id >= total) return;
  int x = (int)(id % h);
  long t = id / h;
  int o = (int)(t % O);
  int b = (int)(t / O);
  float s = 0.f;
  for (int k = 0; k < K; ++k)
    s += mid[(((size_t)b * O + o) * K + k) * h + x];
  out[((size_t)b * O + o) * h + x] = s * (1.f / (float)K);
}

// ============================================================================
// Host side. Input order: JAX pytree flatten of setup_inputs() (sorted keys):
//   gd_float 0..19, gd_int 20..45, params 46..107, pc1=108, pc2=109.
// ============================================================================
enum {
  P_BCN1_B = 46, P_BCN1_W, P_BCN2_B, P_BCN2_W, P_BCN3_B, P_BCN3_W,
  P_BCN4_B, P_BCN4_W, P_BCN5_B, P_BCN5_W,
  P_BCNB1_B, P_BCNB1_W, P_BCNB2_B, P_BCNB2_W, P_BCNB3_B, P_BCNB3_W,
  P_BCNB4_B, P_BCNB4_W, P_BCNB5_B, P_BCNB5_W,
  P_CONV1_B0, P_CONV1_B1, P_CONV1_B2, P_CONV1_W0, P_CONV1_W1, P_CONV1_W2,
  P_CONV2_B, P_CONV2_W, P_CONV3_B, P_CONV3_W, P_CONV4_B, P_CONV4_W,
  P_CORR1_B1, P_CORR1_B2, P_CORR1_W1, P_CORR1_W2,
  P_CORR2_B1, P_CORR2_B2, P_CORR2_W1, P_CORR2_W2,
  P_CORR3_B1, P_CORR3_B2, P_CORR3_W1, P_CORR3_W2,
  P_REF1_B0, P_REF1_B1, P_REF1_B2, P_REF1_W0, P_REF1_W1, P_REF1_W2,
  P_REF2_B0, P_REF2_B1, P_REF2_B2, P_REF2_W0, P_REF2_W1, P_REF2_W2,
  P_REF3_B0, P_REF3_B1, P_REF3_B2, P_REF3_W0, P_REF3_W1, P_REF3_W2,
  IN_PC1, IN_PC2
};

extern "C" void kernel_launch(void* const* d_in, const int* in_sizes, int n_in,
                              void* d_out, int out_size, void* d_ws, size_t ws_size,
                              hipStream_t stream) {
  (void)in_sizes; (void)n_in; (void)out_size; (void)ws_size;
  const int Hs[5]    = {6144, 3072, 1536, 768, 384};
  const int NPTSs[5] = {8192, 6144, 3072, 1536, 768};

  auto fin = [&](int i) { return (const float*)d_in[i]; };
  auto iin = [&](int i) { return (const int*)d_in[i]; };
  auto gdf_bary = [&](int k, int pc) { return fin(k * 4 + (pc ? 2 : 0)); };
  auto gdf_el   = [&](int k, int pc) { return fin(k * 4 + (pc ? 3 : 1)); };
  auto gd_blur  = [&](int k, int pc) { return iin(k < 2 ? 20 + k * 4 + pc * 2     : 28 + (k - 2) * 6 + pc * 3);     };
  auto gd_off   = [&](int k, int pc) { return iin(k < 2 ? 20 + k * 4 + pc * 2 + 1 : 28 + (k - 2) * 6 + pc * 3 + 2); };
  auto gd_corr  = [&](int k, int pc) { return iin(28 + (k - 2) * 6 + pc * 3 + 1); };

  // deterministic bump allocator (floats), 32B-aligned chunks
  float* ws = (float*)d_ws;
  size_t bump = 0;
  auto alloc = [&](size_t nflt) { bump = (bump + 7) & ~(size_t)7; float* p = ws + bump; bump += nflt; return p; };

  float* f1 = alloc(4L * 64 * 8192);
  float* f2 = alloc(4L * 64 * 8192);
  float *Aa[5], *Bb[5];
  for (int k = 0; k < 5; ++k) { Aa[k] = alloc(4L * 64 * Hs[k]); Bb[k] = alloc(4L * 64 * Hs[k]); }
  float* c1b = alloc(4L * 64 * 1536);
  float* c2b = alloc(4L * 64 * 768);
  float* c3b = alloc(4L * 64 * 384);
  float* u5 = alloc(4L * 64 * 768);
  float* u4 = alloc(4L * 64 * 1536);
  float* u3 = alloc(4L * 64 * 3072);
  float* u2 = alloc(4L * 64 * 6144);
  float* u1 = alloc(4L * 128 * 8192);
  float* t0 = alloc(4L * 32 * 8192);   // K-padded copy of pc (3 -> 32 rows)
  float* t1 = alloc(4L * 32 * 8192);
  float* t2 = alloc(4L * 32 * 8192);
  float* LAT   = alloc(4L * 132 * 6144);  // max concat lattice
  float* BLUR  = alloc(4L * 128 * 6144);
  float* PREV  = alloc(4L * 64 * 1536);
  float* HMID  = alloc(4L * 32 * 1536);
  float* CORRB = alloc(4L * 32 * 1536);
  float* R1  = alloc(4L * 64 * 1536);
  float* R2  = alloc(4L * 64 * 1536);
  float* RIN = alloc(4L * 64 * 1536);     // 36 real rows + zero pad to 64
  float* SCRA = alloc(4L * 1024 * 8192);  // padded gathers / corr prod / conv2 out
  float* SCRB = alloc(4L * 512 * 8192);   // corr mid / conv3 out
  __bf16* WP = (__bf16*)alloc(600000);    // packed weights (max 512x1024 bf16)

  auto grid1 = [](long total) { return dim3((unsigned)((total + 255) / 256)); };

  // pack W to bf16 [Opad, Cs] then run WMMA GEMM (X already K-padded to Cs)
  auto gemm = [&](int wi, int bi, const float* X, float* Y,
                  int O, int C, int Cs, int Nn, int act) {
    const int Opad = ((O + 63) / 64) * 64;
    long tp = (long)Opad * Cs;
    hipLaunchKernelGGL(pack_w_kernel, grid1(tp), dim3(256), 0, stream,
                       fin(wi), WP, O, C, Cs, tp);
    dim3 g((Nn + 127) / 128, Opad / 64, NBATCH);
    hipLaunchKernelGGL(gemm_wmma_kernel, g, dim3(256), 0, stream,
                       WP, fin(bi), X, Y, O, Cs, Nn, act);
  };
  auto zero_rows = [&](float* buf, int C, int Cpad, int n) {
    if (Cpad <= C) return;
    long t = 4L * (Cpad - C) * n;
    hipLaunchKernelGGL(zero_rows_kernel, grid1(t), dim3(256), 0, stream,
                       buf, C, Cpad, n, t);
  };

  // ---- stem (3->32->32->64) for both clouds ----
  hipMemsetAsync(t0, 0, (size_t)4 * 32 * 8192 * sizeof(float), stream);
  {
    long tc = 4L * 3 * 8192;
    hipLaunchKernelGGL(copy_ch_kernel, grid1(tc), dim3(256), 0, stream,
                       fin(IN_PC1), t0, 3, 8192, 32, 0, tc);
  }
  gemm(P_CONV1_W0, P_CONV1_B0, t0, t1, 32, 3, 32, 8192, 1);
  gemm(P_CONV1_W1, P_CONV1_B1, t1, t2, 32, 32, 32, 8192, 1);
  gemm(P_CONV1_W2, P_CONV1_B2, t2, f1, 64, 32, 32, 8192, 1);
  {
    long tc = 4L * 3 * 8192;
    hipLaunchKernelGGL(copy_ch_kernel, grid1(tc), dim3(256), 0, stream,
                       fin(IN_PC2), t0, 3, 8192, 32, 0, tc);
  }
  gemm(P_CONV1_W0, P_CONV1_B0, t0, t1, 32, 3, 32, 8192, 1);
  gemm(P_CONV1_W1, P_CONV1_B1, t1, t2, 32, 32, 32, 8192, 1);
  gemm(P_CONV1_W2, P_CONV1_B2, t2, f2, 64, 32, 32, 8192, 1);

  // ---- bcn_splat: splat([el|feat]) -> blur gather (pad 612->640) -> GEMM ----
  auto bcn_splat = [&](int wi, int bi, int k, int pc, const float* feat, float* out) {
    const int h = Hs[k], npt = NPTSs[k];
    hipMemsetAsync(LAT, 0, (size_t)4 * 68 * h * sizeof(float), stream);
    long ts = 4L * 68 * npt;
    hipLaunchKernelGGL(splat_kernel, grid1(ts), dim3(256), 0, stream,
                       gdf_el(k, pc), feat, gdf_bary(k, pc), gd_off(k, pc),
                       LAT, 64, 1, npt, h, ts);
    long tg = 4L * 612 * h;
    hipLaunchKernelGGL(gather_k_kernel, grid1(tg), dim3(256), 0, stream,
                       LAT, gd_blur(k, pc), SCRA, 68, 9, h, 640, tg);
    zero_rows(SCRA, 612, 640, h);
    gemm(wi, bi, SCRA, out, 64, 612, 640, h, 1);
  };

  // ---- corr: prod -> GEMM(32,Ct) over 27*h -> mean -> gather K1 -> GEMM(32,288) ----
  auto corr = [&](int w1i, int b1i, int w2i, int b2i, int k,
                  const float* F1, const float* F2, const float* prev, float* out) {
    const int h = Hs[k], npt = NPTSs[k];
    const int Cp = prev ? 64 : 0, Ct = 64 + Cp;
    if (prev) {
      hipMemsetAsync(PREV, 0, (size_t)4 * 64 * h * sizeof(float), stream);
      long ts = 4L * 64 * npt;
      hipLaunchKernelGGL(splat_kernel, grid1(ts), dim3(256), 0, stream,
                         (const float*)nullptr, prev, gdf_bary(k, 0), gd_off(k, 0),
                         PREV, 64, 0, npt, h, ts);
    }
    long tp = 4L * Ct * 27 * h;
    hipLaunchKernelGGL(corr_prod_kernel, grid1(tp), dim3(256), 0, stream,
                       F1, F2, gd_corr(k, 1), PREV, SCRA, 64, Cp, 27, h, tp);
    gemm(w1i, b1i, SCRA, SCRB, 32, Ct, Ct, 27 * h, 1);
    long tm = 4L * 32 * h;
    hipLaunchKernelGGL(mean_k_kernel, grid1(tm), dim3(256), 0, stream,
                       SCRB, HMID, 32, 27, h, tm);
    long tg = 4L * 288 * h;
    hipLaunchKernelGGL(gather_k_kernel, grid1(tg), dim3(256), 0, stream,
                       HMID, gd_corr(k, 0), SCRA, 32, 9, h, 288, tg);
    gemm(w2i, b2i, SCRA, out, 32, 288, 288, h, 1);
  };

  // ---- refine: optional [el|corr] concat (pad 36->64) then 3 GEMMs ----
  auto refine = [&](int w0, int b0, int w1, int b1, int w2, int b2,
                    const float* el, int h, float* out) {
    const float* in; int ci0, cs0;
    if (el) {
      long tc0 = 4L * 4 * h, tc1 = 4L * 32 * h;
      hipLaunchKernelGGL(copy_ch_kernel, grid1(tc0), dim3(256), 0, stream, el,    RIN, 4,  h, 64, 0, tc0);
      hipLaunchKernelGGL(copy_ch_kernel, grid1(tc1), dim3(256), 0, stream, CORRB, RIN, 32, h, 64, 4, tc1);
      zero_rows(RIN, 36, 64, h);
      in = RIN; ci0 = 36; cs0 = 64;
    } else { in = CORRB; ci0 = 32; cs0 = 32; }
    gemm(w0, b0, in, R1, 64, ci0, cs0, h, 1);
    gemm(w1, b1, R1, R2, 64, 64, 64, h, 1);
    gemm(w2, b2, R2, out, 64, 64, 64, h, 1);
  };

  // ---- bcn_slice: concat lattices -> blur gather (K-padded) -> GEMM -> slice ----
  auto bcn_slice = [&](int wi, int bi, int k, int co,
                       const float* s0, int cc0, const float* s1, int cc1,
                       const float* s2, int cc2, const float* s3, int cc3, float* out) {
    const int h = Hs[k], npt = NPTSs[k];
    const int Ct = cc0 + cc1 + cc2 + cc3;
    const int CK = Ct * 9;
    const int CKp = ((CK + 31) / 32) * 32;
    int ofs = 0;
    const float* srcs[4] = {s0, s1, s2, s3};
    const int    ccs[4]  = {cc0, cc1, cc2, cc3};
    for (int s = 0; s < 4; ++s) {
      if (!srcs[s]) continue;
      long tc = 4L * ccs[s] * h;
      hipLaunchKernelGGL(copy_ch_kernel, grid1(tc), dim3(256), 0, stream,
                         srcs[s], LAT, ccs[s], h, Ct, ofs, tc);
      ofs += ccs[s];
    }
    long tg = 4L * CK * h;
    hipLaunchKernelGGL(gather_k_kernel, grid1(tg), dim3(256), 0, stream,
                       LAT, gd_blur(k, 0), SCRA, Ct, 9, h, CKp, tg);
    zero_rows(SCRA, CK, CKp, h);
    gemm(wi, bi, SCRA, BLUR, co, CK, CKp, h, 1);
    long tsl = 4L * co * npt;
    hipLaunchKernelGGL(slice_kernel, grid1(tsl), dim3(256), 0, stream,
                       BLUR, gdf_bary(k, 0), gd_off(k, 0), out, co, h, npt, tsl);
  };

  // ---- forward graph ----
  bcn_splat(P_BCN1_W, P_BCN1_B, 0, 0, f1, Aa[0]);
  bcn_splat(P_BCN1_W, P_BCN1_B, 0, 1, f2, Bb[0]);
  bcn_splat(P_BCN2_W, P_BCN2_B, 1, 0, Aa[0], Aa[1]);
  bcn_splat(P_BCN2_W, P_BCN2_B, 1, 1, Bb[0], Bb[1]);
  bcn_splat(P_BCN3_W, P_BCN3_B, 2, 0, Aa[1], Aa[2]);
  bcn_splat(P_BCN3_W, P_BCN3_B, 2, 1, Bb[1], Bb[2]);

  corr(P_CORR1_W1, P_CORR1_B1, P_CORR1_W2, P_CORR1_B2, 2, Aa[2], Bb[2], nullptr, CORRB);
  refine(P_REF1_W0, P_REF1_B0, P_REF1_W1, P_REF1_B1, P_REF1_W2, P_REF1_B2,
         gdf_el(3, 0), 1536, c1b);

  bcn_splat(P_BCN4_W, P_BCN4_B, 3, 0, Aa[2], Aa[3]);
  bcn_splat(P_BCN4_W, P_BCN4_B, 3, 1, Bb[2], Bb[3]);
  corr(P_CORR2_W1, P_CORR2_B1, P_CORR2_W2, P_CORR2_B2, 3, Aa[3], Bb[3], c1b, CORRB);
  refine(P_REF2_W0, P_REF2_B0, P_REF2_W1, P_REF2_B1, P_REF2_W2, P_REF2_B2,
         gdf_el(4, 0), 768, c2b);

  bcn_splat(P_BCN5_W, P_BCN5_B, 4, 0, Aa[3], Aa[4]);
  bcn_splat(P_BCN5_W, P_BCN5_B, 4, 1, Bb[3], Bb[4]);
  corr(P_CORR3_W1, P_CORR3_B1, P_CORR3_W2, P_CORR3_B2, 4, Aa[4], Bb[4], c2b, CORRB);
  refine(P_REF3_W0, P_REF3_B0, P_REF3_W1, P_REF3_B1, P_REF3_W2, P_REF3_B2,
         nullptr, 384, c3b);

  bcn_slice(P_BCNB5_W, P_BCNB5_B, 4, 64,  c3b, 64, Aa[4], 64, nullptr, 0, nullptr, 0, u5);
  bcn_slice(P_BCNB4_W, P_BCNB4_B, 3, 64,  gdf_el(4, 0), 4, u5, 64, c2b, 64, Aa[3], 64, u4);
  bcn_slice(P_BCNB3_W, P_BCNB3_B, 2, 64,  gdf_el(3, 0), 4, u4, 64, c1b, 64, Aa[2], 64, u3);
  bcn_slice(P_BCNB2_W, P_BCNB2_B, 1, 64,  gdf_el(2, 0), 4, u3, 64, Aa[1], 64, nullptr, 0, u2);
  bcn_slice(P_BCNB1_W, P_BCNB1_B, 0, 128, gdf_el(1, 0), 4, u2, 64, Aa[0], 64, nullptr, 0, u1);

  // ---- final MLP 128 -> 1024 -> 512 -> 3 ----
  gemm(P_CONV2_W, P_CONV2_B, u1, SCRA, 1024, 128, 128, 8192, 1);
  gemm(P_CONV3_W, P_CONV3_B, SCRA, SCRB, 512, 1024, 1024, 8192, 1);
  gemm(P_CONV4_W, P_CONV4_B, SCRB, (float*)d_out, 3, 512, 512, 8192, 0);
}
